// EvidenceRetriever_88545045775235
// MI455X (gfx1250) — matrix operand order repaired and checked by reading, not compiled, not run on hardware
//
#include <hip/hip_runtime.h>

typedef __bf16 bf16;
typedef __attribute__((ext_vector_type(16))) bf16  v16bf;
typedef __attribute__((ext_vector_type(8)))  float v8f;

#define HID   768
#define BQ    64
#define TOPK  5
#define WAVES 8
#define ITERS 16
#define ROWS_PER_ITER (WAVES * 16)        // 128 evidence rows per block-iteration
#define CHUNK (ROWS_PER_ITER * ITERS)     // 2048 evidence rows per block
#define NEG_INF (-__builtin_inff())

struct Cand { float s; int i; };

// Streaming insert: candidates arrive with strictly ascending indices, so a
// strict '>' comparison reproduces jax top_k tie-breaking (lowest index wins).
__device__ __forceinline__ void insert_stream(float (&S)[TOPK], int (&I)[TOPK],
                                              float s, int idx) {
  if (s > S[TOPK - 1]) {
    S[TOPK - 1] = s; I[TOPK - 1] = idx;
#pragma unroll
    for (int j = TOPK - 1; j > 0; --j) {
      if (S[j] > S[j - 1]) {
        float ts = S[j]; S[j] = S[j - 1]; S[j - 1] = ts;
        int   ti = I[j]; I[j] = I[j - 1]; I[j - 1] = ti;
      }
    }
  }
}

__device__ __forceinline__ bool better(float s1, int i1, float s2, int i2) {
  return (s1 > s2) || (s1 == s2 && i1 < i2);
}

// Comparator insert for merging unordered candidate lists (ties by low index).
__device__ __forceinline__ void insert_cmp(float (&S)[TOPK], int (&I)[TOPK],
                                           float s, int idx) {
  if (better(s, idx, S[TOPK - 1], I[TOPK - 1])) {
    S[TOPK - 1] = s; I[TOPK - 1] = idx;
#pragma unroll
    for (int j = TOPK - 1; j > 0; --j) {
      if (better(S[j], I[j], S[j - 1], I[j - 1])) {
        float ts = S[j]; S[j] = S[j - 1]; S[j - 1] = ts;
        int   ti = I[j]; I[j] = I[j - 1]; I[j - 1] = ti;
      }
    }
  }
}

// ---------------------------------------------------------------------------
// Phase 1: L2-normalize the 64 queries, store bf16 row-major in workspace.
// ---------------------------------------------------------------------------
__global__ void __launch_bounds__(256) qnorm_kernel(const float* __restrict__ q,
                                                    bf16* __restrict__ out) {
  __shared__ float red[256];
  const int row = blockIdx.x;
  const int tid = threadIdx.x;
  const float* r = q + (size_t)row * HID;
  float ss = 0.f;
  for (int i = tid; i < HID; i += 256) { float v = r[i]; ss += v * v; }
  red[tid] = ss;
  __syncthreads();
  for (int s = 128; s > 0; s >>= 1) {
    if (tid < s) red[tid] += red[tid + s];
    __syncthreads();
  }
  const float inv = 1.f / fmaxf(sqrtf(red[0]), 1e-12f);
  for (int i = tid; i < HID; i += 256)
    out[(size_t)row * HID + i] = (bf16)(r[i] * inv);
}

// ---------------------------------------------------------------------------
// Phase 2: single pass over evidence. Per wave: 16 evidence rows (M) x 64
// queries (N), K=768 in steps of 32 via v_wmma_f32_16x16x32_bf16. Evidence
// norm (from the exact f32 data) is fused into the same pass; per-lane top-5
// kept in registers, merged per block into a global candidate buffer.
// ---------------------------------------------------------------------------
__global__ void __launch_bounds__(256) score_topk_kernel(
    const float* __restrict__ evid, const bf16* __restrict__ qn,
    Cand* __restrict__ candOut, int N) {
  extern __shared__ char smem[];
  bf16*  sQ    = (bf16*)smem;                                 // 64*768 bf16 = 96 KB
  Cand*  sCand = (Cand*)(smem + BQ * HID * sizeof(bf16));     // 256*4*5 pairs = 40 KB

  const int tid = threadIdx.x;

  // Stage normalized queries into LDS via async global->LDS DMA.
  // Dynamic LDS starts at offset 0 (no static LDS in this kernel), so the
  // LDS operand is the plain byte offset of the destination.
  {
    const char* gsrc = (const char*)qn;
    for (int i = tid; i < (BQ * HID * 2) / 16; i += 256) {
      unsigned ldsoff = (unsigned)i * 16u;
      const void* g = gsrc + (size_t)i * 16;
      asm volatile("global_load_async_to_lds_b128 %0, %1, off"
                   :: "v"(ldsoff), "v"(g) : "memory");
    }
    asm volatile("s_wait_asynccnt 0x0" ::: "memory");
  }
  __syncthreads();

  const int wv   = tid >> 5;
  const int lane = tid & 31;
  const int lm   = lane & 15;   // row within 16-row tile (A) / query column (B)
  const int hi   = lane >> 4;   // lane-half selector per WMMA layouts

  float topS[4][TOPK]; int topI[4][TOPK];
#pragma unroll
  for (int t = 0; t < 4; ++t)
#pragma unroll
    for (int k = 0; k < TOPK; ++k) { topS[t][k] = NEG_INF; topI[t][k] = 0x7fffffff; }

  const int blockStart = blockIdx.x * CHUNK;

  for (int it = 0; it < ITERS; ++it) {
    const int rowBase = blockStart + it * ROWS_PER_ITER + wv * 16;
    const int row  = rowBase + lm;
    const int rowc = row < N ? row : (N - 1);           // clamp; masked later
    const float* rp = evid + (size_t)rowc * HID;

    // Progressive prefetch of this lane's NEXT-iteration row: one cacheline
    // per K-step covers the full 3 KB row by the time it is consumed,
    // giving deep MLP without consuming LOADcnt (global_prefetch_b8).
    const float* pfp = rp + (size_t)ROWS_PER_ITER * HID;
    const bool   pf  = (row + ROWS_PER_ITER) < N;

    v8f acc0 = {}, acc1 = {}, acc2 = {}, acc3 = {};
    float ss0 = 0.f, ss1 = 0.f, ss2 = 0.f, ss3 = 0.f;

#pragma unroll 4
    for (int k0 = 0; k0 < HID; k0 += 32) {
      if (pf) __builtin_prefetch(pfp + k0, 0, 1);

      // A fragment (contiguous k order, same as B): lanes 0-15 hold K [0,16),
      // lanes 16-31 hold K [16,32); row = lane%16.
      const float4* p0 = (const float4*)(rp + k0 + hi * 16);
      const float4* p1 = (const float4*)(rp + k0 + 8 + hi * 16);
      const float4 a0 = p0[0], a1 = p0[1];
      const float4 b0 = p1[0], b1 = p1[1];

      ss0 = fmaf(a0.x, a0.x, fmaf(a0.y, a0.y, fmaf(a0.z, a0.z, fmaf(a0.w, a0.w, ss0))));
      ss1 = fmaf(a1.x, a1.x, fmaf(a1.y, a1.y, fmaf(a1.z, a1.z, fmaf(a1.w, a1.w, ss1))));
      ss2 = fmaf(b0.x, b0.x, fmaf(b0.y, b0.y, fmaf(b0.z, b0.z, fmaf(b0.w, b0.w, ss2))));
      ss3 = fmaf(b1.x, b1.x, fmaf(b1.y, b1.y, fmaf(b1.z, b1.z, fmaf(b1.w, b1.w, ss3))));

      v16bf af;
      af[0]  = (bf16)a0.x; af[1]  = (bf16)a0.y; af[2]  = (bf16)a0.z; af[3]  = (bf16)a0.w;
      af[4]  = (bf16)a1.x; af[5]  = (bf16)a1.y; af[6]  = (bf16)a1.z; af[7]  = (bf16)a1.w;
      af[8]  = (bf16)b0.x; af[9]  = (bf16)b0.y; af[10] = (bf16)b0.z; af[11] = (bf16)b0.w;
      af[12] = (bf16)b1.x; af[13] = (bf16)b1.y; af[14] = (bf16)b1.z; af[15] = (bf16)b1.w;

      // B fragment (32x16): column = lane%16, lanes 0-15 K=k0..k0+15,
      // lanes 16-31 K=k0+16..k0+31 -> 32 contiguous bytes of bf16 per lane.
      const int kq = k0 + hi * 16;
      const v16bf bq0 = *(const v16bf*)(sQ + (size_t)(lm +  0) * HID + kq);
      const v16bf bq1 = *(const v16bf*)(sQ + (size_t)(lm + 16) * HID + kq);
      const v16bf bq2 = *(const v16bf*)(sQ + (size_t)(lm + 32) * HID + kq);
      const v16bf bq3 = *(const v16bf*)(sQ + (size_t)(lm + 48) * HID + kq);

      acc0 = __builtin_amdgcn_wmma_f32_16x16x32_bf16(false, af, false, bq0, (short)0, acc0, false, false);
      acc1 = __builtin_amdgcn_wmma_f32_16x16x32_bf16(false, af, false, bq1, (short)0, acc1, false, false);
      acc2 = __builtin_amdgcn_wmma_f32_16x16x32_bf16(false, af, false, bq2, (short)0, acc2, false, false);
      acc3 = __builtin_amdgcn_wmma_f32_16x16x32_bf16(false, af, false, bq3, (short)0, acc3, false, false);
    }

    // Lanes l and l+16 each hold half of row (lane%16)'s sum of squares;
    // combine across the pair, then broadcast row norms with shuffles.
    float ss = (ss0 + ss1) + (ss2 + ss3);
    ss += __shfl_xor(ss, 16, 32);
    const float inv = 1.f / fmaxf(sqrtf(ss), 1e-12f);
    float invr[8];
#pragma unroll
    for (int r = 0; r < 8; ++r) invr[r] = __shfl(inv, r + 8 * hi, 32);

    // C layout: VGPR r -> evidence row r + 8*(lane>=16); col = lane%16 (+16*nt).
#pragma unroll
    for (int r = 0; r < 8; ++r) {
      const int idx = rowBase + r + 8 * hi;
      if (idx < N) {
        insert_stream(topS[0], topI[0], acc0[r] * invr[r], idx);
        insert_stream(topS[1], topI[1], acc1[r] * invr[r], idx);
        insert_stream(topS[2], topI[2], acc2[r] * invr[r], idx);
        insert_stream(topS[3], topI[3], acc3[r] * invr[r], idx);
      }
    }
  }

  // Dump per-lane lists to LDS, then one thread per query merges the block.
#pragma unroll
  for (int t = 0; t < 4; ++t)
#pragma unroll
    for (int k = 0; k < TOPK; ++k) {
      Cand c; c.s = topS[t][k]; c.i = topI[t][k];
      sCand[((size_t)tid * 4 + t) * TOPK + k] = c;
    }
  __syncthreads();

  if (tid < BQ) {
    const int q = tid, qm = q & 15, nt = q >> 4;
    float S[TOPK]; int I[TOPK];
#pragma unroll
    for (int k = 0; k < TOPK; ++k) { S[k] = NEG_INF; I[k] = 0x7fffffff; }
    for (int w = 0; w < WAVES; ++w) {
#pragma unroll
      for (int h = 0; h < 2; ++h) {
        const int ln = w * 32 + qm + 16 * h;
        const Cand* c = &sCand[((size_t)ln * 4 + nt) * TOPK];
#pragma unroll
        for (int k = 0; k < TOPK; ++k) insert_cmp(S, I, c[k].s, c[k].i);
      }
    }
    Cand* out = candOut + ((size_t)blockIdx.x * BQ + q) * TOPK;
#pragma unroll
    for (int k = 0; k < TOPK; ++k) { out[k].s = S[k]; out[k].i = I[k]; }
  }
}

// ---------------------------------------------------------------------------
// Phase 3: reduce nblocks*5 candidates per query to the final top-5.
// ---------------------------------------------------------------------------
__global__ void final_topk_kernel(const Cand* __restrict__ cand, int nblocks,
                                  int* __restrict__ outIdx,
                                  float* __restrict__ outScore) {
  const int q = threadIdx.x;
  if (q >= BQ) return;
  float S[TOPK]; int I[TOPK];
#pragma unroll
  for (int k = 0; k < TOPK; ++k) { S[k] = NEG_INF; I[k] = 0x7fffffff; }
  for (int b = 0; b < nblocks; ++b) {
    const Cand* c = cand + ((size_t)b * BQ + q) * TOPK;
#pragma unroll
    for (int k = 0; k < TOPK; ++k) insert_cmp(S, I, c[k].s, c[k].i);
  }
#pragma unroll
  for (int k = 0; k < TOPK; ++k) {
    outIdx[q * TOPK + k]   = I[k];
    outScore[q * TOPK + k] = S[k];
  }
}

extern "C" void kernel_launch(void* const* d_in, const int* in_sizes, int n_in,
                              void* d_out, int out_size, void* d_ws, size_t ws_size,
                              hipStream_t stream) {
  const float* q = (const float*)d_in[0];   // [64, 768] f32
  const float* e = (const float*)d_in[1];   // [N, 768] f32
  (void)n_in; (void)ws_size; (void)out_size;
  const int N = in_sizes[1] / HID;

  // Workspace layout: [0, 96KB) bf16 normalized queries; then candidate pairs.
  bf16* qn = (bf16*)d_ws;
  Cand* cand = (Cand*)((char*)d_ws + (size_t)BQ * HID * sizeof(bf16));

  const int nblocks = (N + CHUNK - 1) / CHUNK;

  // Output: indices (int32) [64*5] then scores (f32) [64*5], flat-concatenated.
  int*   outIdx   = (int*)d_out;
  float* outScore = (float*)d_out + BQ * TOPK;

  qnorm_kernel<<<BQ, 256, 0, stream>>>(q, qn);

  const size_t ldsBytes = (size_t)BQ * HID * sizeof(bf16)        // 98304
                        + (size_t)256 * 4 * TOPK * sizeof(Cand); // 40960
  score_topk_kernel<<<nblocks, 256, ldsBytes, stream>>>(e, qn, cand, N);

  final_topk_kernel<<<1, BQ, 0, stream>>>(cand, nblocks, outIdx, outScore);
}
